// GaussianImage_Cholesky_12618613916324
// MI455X (gfx1250) — matrix operand order repaired and checked by reading, not compiled, not run on hardware
//
#include <hip/hip_runtime.h>
#include <hip/hip_bf16.h>
#include <stdint.h>

// ---------------------------------------------------------------------------
// GaussianImage (Cholesky) sum-rasterizer for MI455X (gfx1250, wave32).
//
// Roofline: ~4e8 pixel-gaussian pairs, each 1 exp + ~10 VALU -> compute bound
// (output is only 2.4 MB; HBM at 23.3 TB/s is irrelevant). The color
// accumulation einsum('nhw,nc->hwc') is a GEMM (M=HW, K=N, N=3) mapped onto
// v_wmma_f32_16x16x32_f16. Param staging uses the CDNA5 Tensor Data Mover
// (tensor_load_to_lds, TENSORcnt) with LDS double buffering so the HBM/L2
// fetch of gaussian params overlaps the exp/WMMA work; cooperative-load
// fallback if the TDM builtin is unavailable.
// ---------------------------------------------------------------------------

#define T_FRAMES 3
#define NGAUSS   2000
#define NPAD     2048          // ceil(N/256)*256, zero padded
#define IMG_H    256
#define IMG_W    256
#define DEG      2
#define CHUNK_K  32            // K per WMMA
#define TILE_K   256           // gaussians staged in LDS per outer iteration
#define NTILES   (NPAD / TILE_K)    // 8
#define NCHUNKS  (NPAD / CHUNK_K)   // 64

typedef __attribute__((ext_vector_type(16))) _Float16     v16h;
typedef __attribute__((ext_vector_type(8)))  float        v8f;
typedef __attribute__((ext_vector_type(4)))  unsigned int v4u;
typedef __attribute__((ext_vector_type(4)))  int          v4i;
typedef __attribute__((ext_vector_type(8)))  int          v8i_t;

#if defined(__has_builtin)
#if __has_builtin(__builtin_amdgcn_tensor_load_to_lds) && \
    __has_builtin(__builtin_amdgcn_s_wait_tensorcnt)
#define HAVE_TDM 1
#endif
#endif
#ifndef HAVE_TDM
#define HAVE_TDM 0
#endif

__device__ __forceinline__ float sigmoidf_(float x) {
    return 1.0f / (1.0f + __expf(-x));
}

// ---------------------------------------------------------------------------
// Kernel 1: per (frame, gaussian) parameters -> 8-float record
//   rec = { mx, my, conicA, conicB, conicC, opac, 0, 0 }
// Padding records (n >= NGAUSS) are all-zero: alpha = 0*exp(0) = 0 (no NaN).
// ---------------------------------------------------------------------------
__global__ __launch_bounds__(256) void k_params(
    const float* __restrict__ xyz,   // [DEG+1, N, 2]
    const float* __restrict__ chol,  // [DEG+1, N, 3]
    const float* __restrict__ opac,  // [DEG+1, N, 1]
    float* __restrict__ P)           // [T, NPAD, 8]
{
    int idx = blockIdx.x * blockDim.x + threadIdx.x;
    if (idx >= T_FRAMES * NPAD) return;
    int t = idx / NPAD;
    int n = idx % NPAD;
    float* rec = P + (size_t)(t * NPAD + n) * 8;
    if (n >= NGAUSS) {
        #pragma unroll
        for (int i = 0; i < 8; ++i) rec[i] = 0.0f;
        return;
    }
    float tv = (float)t / (float)(T_FRAMES - 1);
    float tp[DEG + 1];
    tp[0] = 1.0f;
    #pragma unroll
    for (int d = 1; d <= DEG; ++d) tp[d] = tp[d - 1] * tv;

    float px = 0.f, py = 0.f, l1 = 0.f, l2 = 0.f, l3 = 0.f, ov = 0.f;
    #pragma unroll
    for (int d = 0; d <= DEG; ++d) {
        px += xyz[(size_t)d * NGAUSS * 2 + n * 2 + 0] * tp[d];
        py += xyz[(size_t)d * NGAUSS * 2 + n * 2 + 1] * tp[d];
        l1 += chol[(size_t)d * NGAUSS * 3 + n * 3 + 0] * tp[d];
        l2 += chol[(size_t)d * NGAUSS * 3 + n * 3 + 1] * tp[d];
        l3 += chol[(size_t)d * NGAUSS * 3 + n * 3 + 2] * tp[d];
        ov += opac[(size_t)d * NGAUSS + n] * tp[d];
    }
    l1 += 0.5f;  // CHOL_BOUND = [0.5, 0.0, 0.5]
    l3 += 0.5f;

    float mx = 0.5f * (float)IMG_W * (tanhf(px) + 1.0f);
    float my = 0.5f * (float)IMG_H * (tanhf(py) + 1.0f);
    float a = l1 * l1;
    float b = l1 * l2;
    float c = l2 * l2 + l3 * l3;
    float inv = 1.0f / (a * c - b * b);   // det = (l1*l3)^2 > 0

    rec[0] = mx;
    rec[1] = my;
    rec[2] = c * inv;      // conic a
    rec[3] = -b * inv;     // conic b
    rec[4] = a * inv;      // conic c
    rec[5] = sigmoidf_(ov);
    rec[6] = 0.0f;
    rec[7] = 0.0f;
}

// ---------------------------------------------------------------------------
// Kernel 2: precompute WMMA B-fragments (colors, f16) per 32-gaussian chunk.
// B (32x16) k order chosen here (not AMD's table order): lane holds column n = lane&15;
// lanes 0-15 hold K=0..15, lanes 16-31 hold K=16..31 (elements 0..15).
// Columns n >= 3 are zero.
// ---------------------------------------------------------------------------
__global__ __launch_bounds__(256) void k_bfrag(
    const float* __restrict__ fdc,   // [N, 3]
    _Float16* __restrict__ B)        // [NCHUNKS, 32 lanes, 16 halves]
{
    int idx = blockIdx.x * blockDim.x + threadIdx.x;
    if (idx >= NCHUNKS * 32) return;
    int chunk = idx >> 5;
    int lane  = idx & 31;
    int n     = lane & 15;
    int kbase = chunk * CHUNK_K + (lane >> 4) * 16;
    v16h v;
    #pragma unroll
    for (int j = 0; j < 16; ++j) {
        int k = kbase + j;
        float c = 0.0f;
        if (n < 3 && k < NGAUSS) c = sigmoidf_(fdc[k * 3 + n]);
        v[j] = (_Float16)c;
    }
    *(v16h*)(B + (size_t)idx * 16) = v;
}

// ---------------------------------------------------------------------------
// TDM: DMA one 8 KB param tile (1024 x 8-byte elements, 1D) into LDS.
// D# per CDNA5 ISA:
//  Group0: w0 = count=1; w1 = lds_addr; w2 = global_addr[31:0];
//          w3 = global_addr[56:32] | type(2)<<30
//  Group1: w0 = data_size(3 -> 8B) << 16
//          tensor_dim0 = 1024 (w1[31:16] lo, w2[15:0] hi)
//          tensor_dim1 = 1    (w2[31:16] lo, w3[15:0] hi)
//          tile_dim0   = 1024 (w3[31:16]); tile_dim1 = 1 (w4[15:0])
//          tensor_dim0_stride = 1024 (w5, w6[15:0])
//  Groups 2/3: zero (<=2D tensor).
// ---------------------------------------------------------------------------
#if HAVE_TDM
__device__ __forceinline__ void tdm_load_tile(const float* gsrc,
                                              unsigned int lds_addr)
{
    unsigned long long ga = (unsigned long long)(uintptr_t)gsrc;
    v4u g0;
    g0[0] = 1u;                                            // count=1
    g0[1] = lds_addr;                                      // lds_addr
    g0[2] = (unsigned int)ga;                              // global_addr lo
    g0[3] = (unsigned int)((ga >> 32) & 0x1FFFFFFull)      // global_addr hi
          | (2u << 30);                                    // type = 2 (image)
    v8i_t g1;
    g1[0] = (int)(3u << 16);          // data_size = 3 (8 bytes)
    g1[1] = (int)(1024u << 16);       // tensor_dim0[15:0]
    g1[2] = (int)(1u << 16);          // tensor_dim0[31:16]=0 | tensor_dim1 lo=1
    g1[3] = (int)(1024u << 16);       // tensor_dim1 hi=0 | tile_dim0=1024
    g1[4] = 1;                        // tile_dim1=1, tile_dim2=0
    g1[5] = 1024;                     // tensor_dim0_stride lo
    g1[6] = 0;                        // stride hi / dim1_stride lo
    g1[7] = 0;
    v4i z4 = {0, 0, 0, 0};
#if defined(__clang_major__) && (__clang_major__ >= 23)
    v8i_t z8 = {0, 0, 0, 0, 0, 0, 0, 0};
    __builtin_amdgcn_tensor_load_to_lds(g0, g1, z4, z4, z8, 0);
#else
    __builtin_amdgcn_tensor_load_to_lds(g0, g1, z4, z4, 0);
#endif
}
#endif

// ---------------------------------------------------------------------------
// Kernel 3: render. One wave (32 lanes) owns a 16x1 pixel strip of one frame.
// K loop over NPAD gaussians: TDM double-buffers 256-gaussian param tiles in
// LDS (wave 0 issues DMA for tile i+1 while all waves compute tile i), then
// 8 chunks of K=32, each closing with one v_wmma_f32_16x16x32_f16.
//
// A (16x32) k order chosen here: lane&15 = pixel row M; lane<16 owns K 0-15,
// lane>=16 owns K 16-31 (matches B fragments),
//   element j : k = (lane>>4)*16 + j
// koff = (lane>>4)*8 is used for the D rows.
// Half-waves read identical LDS addresses -> broadcast, conflict free.
// D layout: channel = lane&15; row M = koff + r for accumulator VGPR r.
// ---------------------------------------------------------------------------
__global__ __launch_bounds__(256) void k_render(
    const float*    __restrict__ P,    // [T, NPAD, 8]
    const _Float16* __restrict__ B,    // [NCHUNKS, 32, 16]
    float* __restrict__ out)           // [T, 3, H, W]
{
    __shared__ float sP[2 * TILE_K * 8];   // two 8 KB tiles (double buffer)

    int tid  = threadIdx.x;
    int wave = tid >> 5;
    int lane = tid & 31;

    const int groupsPerFrame = IMG_H * (IMG_W / 16);       // 4096
    const int blocksPerFrame = groupsPerFrame / 8;         // 512
    int t     = blockIdx.x / blocksPerFrame;
    int bin   = blockIdx.x % blocksPerFrame;
    int group = bin * 8 + wave;                            // 0..4095
    int y     = group >> 4;
    int x0    = (group & 15) << 4;

    int   m    = lane & 15;            // pixel index within strip (A row)
    int   koff = (lane >> 4) * 8;
    float pxf  = (float)(x0 + m);
    float pyf  = (float)y;

    const float* Pt = P + (size_t)t * NPAD * 8;
    v8f acc = {};

#if HAVE_TDM
    // LDS byte offsets of the two buffers (generic ptr low 32 bits == LDS addr)
    unsigned int ldsOff0 = (unsigned int)(uintptr_t)(void*)&sP[0];
    unsigned int ldsOff1 = (unsigned int)(uintptr_t)(void*)&sP[TILE_K * 8];

    if (wave == 0) {                       // prologue: DMA tile 0 -> buf 0
        tdm_load_tile(Pt, ldsOff0);
        __builtin_amdgcn_s_wait_tensorcnt(0);
    }
    __syncthreads();

    for (int it = 0; it < NTILES; ++it) {
        // Issue DMA for tile it+1 into the other buffer. That buffer was last
        // read in iteration it-1; the barrier ending it-1 ordered those reads
        // before this issue.
        if (wave == 0 && (it + 1) < NTILES) {
            tdm_load_tile(Pt + (size_t)(it + 1) * TILE_K * 8,
                          ((it + 1) & 1) ? ldsOff1 : ldsOff0);
        }
        const float* sbuf = sP + (it & 1) * (TILE_K * 8);
#else
    for (int it = 0; it < NTILES; ++it) {
        int kk = it * TILE_K;
        __syncthreads();
        {   // cooperative stage: thread tid loads gaussian (kk + tid)
            const float4* src = (const float4*)(Pt + (size_t)(kk + tid) * 8);
            float4 p0 = src[0];
            float4 p1 = src[1];
            float4* d = (float4*)(sP + tid * 8);
            d[0] = p0;
            d[1] = p1;
        }
        if (kk + TILE_K < NPAD) {      // gfx1250 global_prefetch_b8 of next tile
            __builtin_prefetch(Pt + (size_t)(kk + TILE_K + tid) * 8, 0, 0);
        }
        __syncthreads();
        const float* sbuf = sP;
#endif

        #pragma unroll
        for (int c8 = 0; c8 < TILE_K / CHUNK_K; ++c8) {
            int cb      = c8 * CHUNK_K;
            int chunkId = it * (TILE_K / CHUNK_K) + c8;

            // B fragment: one coalesced 32B load per lane
            v16h bf = *(const v16h*)(B + (size_t)chunkId * 512 + lane * 16);

            // A fragment: 16 alpha evaluations per lane
            v16h af;
            #pragma unroll
            for (int j = 0; j < 16; ++j) {
                int k = cb + (lane >> 4) * 16 + j;
                const float4* g = (const float4*)(sbuf + k * 8);
                float4 g0 = g[0];
                float4 g1 = g[1];
                float dx = g0.x - pxf;
                float dy = g0.y - pyf;
                float q  = g0.z * dx * dx + 2.0f * g0.w * dx * dy + g1.x * dy * dy;
                float al = g1.y * __expf(-0.5f * q);
                al = fminf(0.999f, al);                  // ALPHA_CLAMP
                al = (al < (1.0f / 255.0f)) ? 0.0f : al; // ALPHA_THRESH skip
                af[j] = (_Float16)al;
            }

            // D = A x B + C  (wave-uniform control flow -> EXEC all ones)
            acc = __builtin_amdgcn_wmma_f32_16x16x32_f16(
                false, af, false, bf, (short)0, acc, false, false);
        }

#if HAVE_TDM
        // Ensure tile it+1 has fully landed in LDS before anyone reads it.
        if (wave == 0 && (it + 1) < NTILES) {
            __builtin_amdgcn_s_wait_tensorcnt(0);
        }
        __syncthreads();
#endif
    }

    // Writeback: lane column = channel; rows M = koff + r
    int ch = lane & 15;
    if (ch < 3) {
        float* o = out + ((size_t)t * 3 + ch) * IMG_H * IMG_W
                       + (size_t)y * IMG_W + x0;
        #pragma unroll
        for (int r = 0; r < 8; ++r) {
            float v = acc[r];
            v = fminf(fmaxf(v, 0.0f), 1.0f);   // clip(img, 0, 1)
            o[koff + r] = v;
        }
    }
}

// ---------------------------------------------------------------------------
extern "C" void kernel_launch(void* const* d_in, const int* in_sizes, int n_in,
                              void* d_out, int out_size, void* d_ws, size_t ws_size,
                              hipStream_t stream)
{
    const float* xyz  = (const float*)d_in[0];  // [DEG+1, N, 2]
    const float* chol = (const float*)d_in[1];  // [DEG+1, N, 3]
    const float* opac = (const float*)d_in[2];  // [DEG+1, N, 1]
    const float* fdc  = (const float*)d_in[3];  // [N, 3]
    float* out = (float*)d_out;                 // [T, 3, H, W]

    float*    P = (float*)d_ws;                                    // 192 KB
    _Float16* B = (_Float16*)((char*)d_ws +
                  (size_t)T_FRAMES * NPAD * 8 * sizeof(float));    // +64 KB

    int n1 = T_FRAMES * NPAD;
    k_params<<<(n1 + 255) / 256, 256, 0, stream>>>(xyz, chol, opac, P);

    int n2 = NCHUNKS * 32;
    k_bfrag<<<(n2 + 255) / 256, 256, 0, stream>>>(fdc, B);

    const int blocksPerFrame = (IMG_H * (IMG_W / 16)) / 8;  // 512
    k_render<<<T_FRAMES * blocksPerFrame, 256, 0, stream>>>(P, B, out);

    (void)in_sizes; (void)n_in; (void)out_size; (void)ws_size;
}
